// gate_86019605004293
// MI455X (gfx1250) — matrix operand order repaired and checked by reading, not compiled, not run on hardware
//
#include <hip/hip_runtime.h>

// Problem constants (from reference): B=8, W=512, F=32, P=128
#define Bdim 8
#define Wdim 512
#define Fdim 32
#define Pdim 128
#define NTHREADS 256
#define NWAVES 8

typedef _Float16 v16h __attribute__((ext_vector_type(16)));
typedef float    v8f  __attribute__((ext_vector_type(8)));

// ---- dynamic LDS layout (bytes) ----
// xpadH : 640 f16                 [0,      1280)
// WqH   : 128*128 f16             [1280,   34048)
// WkH   : 128*128 f16             [34048,  66816)
// kH    : 512*128 f16             [66816,  197888)
// qscA  : 8 waves * 16*128 f16    [197888, 230656)
// vL    : 512 f32                 [230656, 232704)
// bqL   : 128 f32                 [232704, 233216)
// bkL   : 128 f32                 [233216, 233728)
// invsL : 512 f32                 [233728, 235776)
// inv2L : 512 f32                 [235776, 237824)
// ZL    : 512 f32                 [237824, 239872)
#define SMEM_BYTES 239872

__device__ __forceinline__ v8f zero8() {
  v8f z = {0.f, 0.f, 0.f, 0.f, 0.f, 0.f, 0.f, 0.f};
  return z;
}

__device__ __forceinline__ v8f wmma16(v16h a, v16h b, v8f c) {
  // D = A(16x32 f16) x B(32x16 f16) + C(16x16 f32)
  return __builtin_amdgcn_wmma_f32_16x16x32_f16(false, a, false, b, (short)0, c,
                                                false, false);
}

// A-fragment (16x32 f16) for the sliding-window matrix:
// A[m][k] = xpad[w0 + m + p0 + k].  ISA layout: lane = m + 16*((k>>3)&1),
// elements 0..7 -> k = 8*hi + 0..7, elements 8..15 -> k = 16 + 8*hi + 0..7.
__device__ __forceinline__ v16h load_a_window(const _Float16* xp, int w0, int p0,
                                              int lane) {
  const int m = lane & 15, hi = lane >> 4;
  const _Float16* p = xp + w0 + m + p0 + 8 * hi;
  v16h a;
#pragma unroll
  for (int t = 0; t < 8; ++t) { a[t] = p[t]; a[8 + t] = p[16 + t]; }
  return a;
}

// A-fragment from a row-major [16][128] f16 tile: A[m][k] = q[m*128 + q0 + k]
__device__ __forceinline__ v16h load_a_rm(const _Float16* q, int q0, int lane) {
  const int m = lane & 15, hi = lane >> 4;
  const _Float16* p = q + m * Pdim + q0 + 8 * hi;
  v16h a;
#pragma unroll
  for (int t = 0; t < 8; ++t) { a[t] = p[t]; a[8 + t] = p[16 + t]; }
  return a;
}

// B-fragment (32x16 f16) from a row-major [rows][128] f16 matrix M:
// B[k][n] = M[row0+n][col0+k].  ISA layout: lane = n + 16*((k>>3)&1),
// elements 0..7 -> k = 8*hi + 0..7, elements 8..15 -> k = 16 + 8*hi + 0..7.
__device__ __forceinline__ v16h load_b_rm(const _Float16* base, int row0, int col0,
                                          int lane) {
  const int n = lane & 15, kg = lane >> 4;
  const _Float16* p = base + (row0 + n) * Pdim + col0 + 8 * kg;
  v16h b;
#pragma unroll
  for (int t = 0; t < 8; ++t) { b[t] = p[t]; b[8 + t] = p[16 + t]; }
  return b;
}

__global__ __launch_bounds__(NTHREADS)
void attn_head_kernel(const float* __restrict__ x, const float* __restrict__ Wq,
                      const float* __restrict__ Wk, const float* __restrict__ bq,
                      const float* __restrict__ bk, const float* __restrict__ sigmas,
                      const float* __restrict__ scaler, const float* __restrict__ vbias,
                      float* __restrict__ out_res, float* __restrict__ out_att) {
  extern __shared__ char smem[];
  _Float16* xpadH = (_Float16*)(smem + 0);
  _Float16* WqH   = (_Float16*)(smem + 1280);
  _Float16* WkH   = (_Float16*)(smem + 34048);
  _Float16* kH    = (_Float16*)(smem + 66816);
  _Float16* qscA  = (_Float16*)(smem + 197888);
  float* vL    = (float*)(smem + 230656);
  float* bqL   = (float*)(smem + 232704);
  float* bkL   = (float*)(smem + 233216);
  float* invsL = (float*)(smem + 233728);
  float* inv2L = (float*)(smem + 235776);
  float* ZL    = (float*)(smem + 237824);

  const int head = blockIdx.x;          // head = b*F + f (matches att layout)
  const int b = head / Fdim, f = head % Fdim;
  const int tid = threadIdx.x;
  const int wave = tid >> 5, lane = tid & 31;
  const int n = lane & 15, hi = lane >> 4;

  const float scl = scaler[f];
  const float vbs = vbias[f];

  // ---------------- phase 0: stage head data into LDS ----------------
  for (int j = tid; j < Wdim + Pdim; j += NTHREADS) {
    const int ws = (j < Pdim) ? j : (j - Pdim);   // xpad = [xt[:, :P], xt]
    xpadH[j] = (_Float16)x[((size_t)b * Wdim + ws) * Fdim + f];
  }
  for (int w = tid; w < Wdim; w += NTHREADS) {
    const float xv = x[((size_t)b * Wdim + w) * Fdim + f];
    vL[w] = scl * xv + vbs;                       // v = scaler*val + vbias
    const float s = sigmas[w];
    invsL[w] = 1.0f / s;
    inv2L[w] = 1.0f / (2.0f * s * s);
  }
  if (tid < Pdim) {
    bqL[tid] = bq[f * Pdim + tid];
    bkL[tid] = bk[f * Pdim + tid];
  }
  for (int idx = tid; idx < Pdim * Pdim; idx += NTHREADS) {
    WqH[idx] = (_Float16)Wq[(size_t)f * Pdim * Pdim + idx];  // row-major [qd][p]
    WkH[idx] = (_Float16)Wk[(size_t)f * Pdim * Pdim + idx];
  }
  __syncthreads();

  // Z[i] = sum_m exp(-(m-i)^2 / 2sig_m^2) / sig_m   (gaussian_curve column sums)
  for (int i = tid; i < Wdim; i += NTHREADS) {
    float z = 0.f;
    for (int m = 0; m < Wdim; ++m) {
      const float d = (float)(m - i);
      z += __expf(-d * d * inv2L[m]) * invsL[m];
    }
    ZL[i] = z;
  }

  // ---------------- phase 1: K = qk_in @ Wk^T + bk  ->  kH (f16) ----------------
  for (int t = wave; t < (Wdim / 16) * (Pdim / 16); t += NWAVES) {
    const int w0 = (t >> 3) * 16, qd0 = (t & 7) * 16;
    v8f acc = zero8();
#pragma unroll
    for (int p0 = 0; p0 < Pdim; p0 += 32) {
      const v16h a  = load_a_window(xpadH, w0, p0, lane);
      const v16h bb = load_b_rm(WkH, qd0, p0, lane);
      acc = wmma16(a, bb, acc);
    }
    const float bias = bkL[qd0 + n];
#pragma unroll
    for (int r = 0; r < 8; ++r)
      kH[(w0 + r + 8 * hi) * Pdim + qd0 + n] = (_Float16)(acc[r] + bias);
  }
  __syncthreads();   // kH + ZL ready for all waves

  // ---------------- phase 2: per 16-row i-tile, fused attention ----------------
  _Float16* qsc = qscA + wave * 16 * Pdim;          // private per-wave q-tile
  const float qscale = 0.08838834764831843f;        // 1/sqrt(128)
  float* attHead = out_att + (size_t)head * Wdim * Wdim;

  for (int it = wave; it < Wdim / 16; it += NWAVES) {
    const int i0 = it * 16;

    // 2a: q tile (16x128) = (window @ Wq^T + bq) * qscale, stored f16 in LDS
    for (int qdt = 0; qdt < 8; ++qdt) {
      const int qd0 = qdt * 16;
      v8f acc = zero8();
#pragma unroll
      for (int p0 = 0; p0 < Pdim; p0 += 32) {
        const v16h a  = load_a_window(xpadH, i0, p0, lane);
        const v16h bb = load_b_rm(WqH, qd0, p0, lane);
        acc = wmma16(a, bb, acc);
      }
      const float bias = bqL[qd0 + n];
#pragma unroll
      for (int r = 0; r < 8; ++r)
        qsc[(r + 8 * hi) * Pdim + qd0 + n] = (_Float16)((acc[r] + bias) * qscale);
    }

    // 2b: keep the q-tile A-fragments (K = 0..127) resident in VGPRs
    v16h aq[4];
#pragma unroll
    for (int c4 = 0; c4 < 4; ++c4) aq[c4] = load_a_rm(qsc, c4 * 32, lane);

    // pass 1: per-lane online row max / sum(exp) over all 512 columns
    float lm[8], ls[8];
#pragma unroll
    for (int r = 0; r < 8; ++r) { lm[r] = -3.0e38f; ls[r] = 0.f; }
    for (int jt = 0; jt < Wdim / 16; ++jt) {
      const int j0 = jt * 16;
      v8f acc = zero8();
#pragma unroll
      for (int c4 = 0; c4 < 4; ++c4) {
        const v16h bb = load_b_rm(kH, j0, c4 * 32, lane);
        acc = wmma16(aq[c4], bb, acc);
      }
#pragma unroll
      for (int r = 0; r < 8; ++r) {
        const float vsc = acc[r];
        const float nm = fmaxf(lm[r], vsc);
        ls[r] = ls[r] * __expf(lm[r] - nm) + __expf(vsc - nm);
        lm[r] = nm;
      }
    }
    // combine the 16 lanes of each half-wave (each holds 1/16 of the columns)
    float rowm[8], rinv[8], zinv[8];
#pragma unroll
    for (int r = 0; r < 8; ++r) {
      float gm = lm[r];
#pragma unroll
      for (int msk = 1; msk < 16; msk <<= 1)
        gm = fmaxf(gm, __shfl_xor(gm, msk, 32));
      float s2 = ls[r] * __expf(lm[r] - gm);
#pragma unroll
      for (int msk = 1; msk < 16; msk <<= 1) s2 += __shfl_xor(s2, msk, 32);
      rowm[r] = gm;
      rinv[r] = 1.0f / s2;
      zinv[r] = 1.0f / ZL[i0 + r + 8 * hi];
    }

    // pass 2: recompute scores, emit att_save (NT), fold curve, accumulate res.
    // att2 logits = att_save*(1-curve) are in [0,1): no max-sub needed.
    float rn[8], rd[8];
#pragma unroll
    for (int r = 0; r < 8; ++r) { rn[r] = 0.f; rd[r] = 0.f; }
    for (int jt = 0; jt < Wdim / 16; ++jt) {
      const int j0 = jt * 16;
      v8f acc = zero8();
#pragma unroll
      for (int c4 = 0; c4 < 4; ++c4) {
        const v16h bb = load_b_rm(kH, j0, c4 * 32, lane);
        acc = wmma16(aq[c4], bb, acc);
      }
      const int j = j0 + n;
      const float ci2 = inv2L[j];
      const float cis = invsL[j];
      const float vj  = vL[j];
#pragma unroll
      for (int r = 0; r < 8; ++r) {
        const int i = i0 + r + 8 * hi;
        const float p = __expf(acc[r] - rowm[r]) * rinv[r];   // att_save[i][j]
        __builtin_nontemporal_store(p, attHead + (size_t)i * Wdim + j);
        const float d = (float)(i - j);
        const float h = __expf(-d * d * ci2) * cis * zinv[r]; // curve[i][j]
        const float e2 = __expf(p * (1.0f - h));
        rd[r] += e2;
        rn[r] += e2 * vj;
      }
    }
    // final cross-lane reduction and res write (res.T layout: out[b][i][f])
#pragma unroll
    for (int r = 0; r < 8; ++r) {
      float den = rd[r], num = rn[r];
#pragma unroll
      for (int msk = 1; msk < 16; msk <<= 1) {
        den += __shfl_xor(den, msk, 32);
        num += __shfl_xor(num, msk, 32);
      }
      if (n == 0) {
        const int i = i0 + r + 8 * hi;
        out_res[((size_t)b * Wdim + i) * Fdim + f] = num / den;
      }
    }
  }
}

extern "C" void kernel_launch(void* const* d_in, const int* in_sizes, int n_in,
                              void* d_out, int out_size, void* d_ws, size_t ws_size,
                              hipStream_t stream) {
  (void)in_sizes; (void)n_in; (void)d_ws; (void)ws_size; (void)out_size;
  // setup_inputs order: x, ratios(unused), Wq, Wk, bq, bk, sigmas, scaler, vbias
  const float* x      = (const float*)d_in[0];
  const float* Wq     = (const float*)d_in[2];
  const float* Wk     = (const float*)d_in[3];
  const float* bq     = (const float*)d_in[4];
  const float* bk     = (const float*)d_in[5];
  const float* sigmas = (const float*)d_in[6];
  const float* scaler = (const float*)d_in[7];
  const float* vbias  = (const float*)d_in[8];

  float* out_res = (float*)d_out;                               // (B,W,F)
  float* out_att = out_res + (size_t)Bdim * Wdim * Fdim;        // (B,F,W,W)

  // Allow >64KB dynamic LDS (gfx1250 WGP has 320KB). Deterministic, no state.
  (void)hipFuncSetAttribute((const void*)attn_head_kernel,
                            hipFuncAttributeMaxDynamicSharedMemorySize,
                            SMEM_BYTES);

  attn_head_kernel<<<dim3(Bdim * Fdim), dim3(NTHREADS), SMEM_BYTES, stream>>>(
      x, Wq, Wk, bq, bk, sigmas, scaler, vbias, out_res, out_att);
}
